// MVP_CrossAttn_EnvGridPlus_64759516889854
// MI455X (gfx1250) — matrix operand order repaired and checked by reading, not compiled, not run on hardware
//
#include <hip/hip_runtime.h>
#include <hip/hip_bf16.h>
#include <math.h>

typedef __attribute__((ext_vector_type(16))) _Float16 v16h;
typedef __attribute__((ext_vector_type(8)))  float    v8f;

constexpr int CB = 8;      // batch
constexpr int CN = 256;    // meas tokens
constexpr int CM = 4;      // bs tokens
constexpr int CS = 260;    // S = N + M
constexpr int CSP = 272;   // S padded to 16
constexpr int CSPK = 288;  // S padded to 32 (prob k-dim)
constexpr int CK = 4096;   // grid tokens (64x64)
constexpr int CD = 128;    // model dim
constexpr int NH = 4;      // heads
constexpr int CDH = 32;    // head dim
constexpr int CE = 32;     // env channels
constexpr int CD4 = 512;   // ff hidden

__device__ __forceinline__ float gelu_f(float x) {
  return 0.5f * x * (1.0f + erff(x * 0.70710678118654752f));
}

// ---------------- WMMA fragment helpers (CDNA5 16x16x32 f16) ----------------
// A 16x32: lanes 0-15 M=0..15, VGPR r<4 -> K=2r,2r+1 ; r>=4 -> K=16+2(r-4)
//          lanes 16-31 same M, K offset +8 (low half) / +8 in upper block.
__device__ __forceinline__ v16h load_a_frag(const _Float16* p0, int ld, int lane) {
  const int m = lane & 15;
  const int koff = (lane >> 4) << 3;  // 0 or 8
  const _Float16* p = p0 + (size_t)m * ld;
  v16h a;
#pragma unroll
  for (int r = 0; r < 8; ++r) {
    int k = ((r < 4) ? 0 : 16) + koff + ((r & 3) << 1);
    a[2 * r]     = p[k];
    a[2 * r + 1] = p[k + 1];
  }
  return a;
}
// B 32x16 from Wt[N][Kd] row-major: lane n = lane&15, same K layout as A (lanes>=16 K offset +8).
__device__ __forceinline__ v16h load_b_frag(const _Float16* p0, int ld, int lane) {
  const int n = lane & 15;
  const int kb = (lane >> 4) << 3;  // 0 or 8
  const _Float16* p = p0 + (size_t)n * ld + kb;
  v16h b;
#pragma unroll
  for (int r = 0; r < 16; ++r) b[r] = p[(r < 8) ? r : (r + 8)];
  return b;
}
__device__ __forceinline__ v8f wmma_f16(v16h a, v16h b, v8f c) {
  return __builtin_amdgcn_wmma_f32_16x16x32_f16(false, a, false, b, (short)0, c,
                                                false, false);
}

// ---------------- generic fused GEMM: C = act(A*Wt^T + bias + epilogue) -----
__device__ __forceinline__ void store_tile(
    v8f acc, int row0, int col0, int lane, int ldc, const float* bias,
    const float* resid, const float* rowscale, const float* colvec,
    const float* colbias, float* Cf, _Float16* Ch, int act) {
  const int n = col0 + (lane & 15);
  const int mb = row0 + ((lane >> 4) << 3);
  const float bn = bias ? bias[n] : 0.f;
  const float cv = colvec ? colvec[n] : 0.f;
  const float cb = colbias ? colbias[n] : 0.f;
#pragma unroll
  for (int r = 0; r < 8; ++r) {
    const int m = mb + r;
    const size_t idx = (size_t)m * ldc + n;
    float v = acc[r] + bn;
    if (rowscale) v += rowscale[m] * cv + cb;
    if (resid) v += resid[idx];
    if (act == 1) v = gelu_f(v);
    if (Cf) Cf[idx] = v;
    if (Ch) Ch[idx] = (_Float16)v;
  }
}

__global__ __launch_bounds__(256) void k_gemm(
    const _Float16* __restrict__ A, const _Float16* __restrict__ Wt,
    const float* __restrict__ bias, const float* __restrict__ resid,
    const float* __restrict__ rowscale, const float* __restrict__ colvec,
    const float* __restrict__ colbias, float* __restrict__ Cf,
    _Float16* __restrict__ Ch, int Mrows, int Ndim, int Kd, int act,
    const int* __restrict__ task_id, int rows_per_task) {
  const int lane = threadIdx.x;
  const int ntg = Ndim >> 6;  // groups of 64 cols
  const int tile = blockIdx.x * 8 + threadIdx.y;
  const int mt = tile / ntg;
  if (mt * 16 >= Mrows) return;
  const int ng = tile - mt * ntg;
  const _Float16* Wp = Wt;
  const float* bp = bias;
  if (task_id) {
    const int tsk = task_id[(mt * 16) / rows_per_task];
    Wp += (size_t)tsk * Ndim * Kd;
    bp += (size_t)tsk * Ndim;
  }
  v8f acc[4] = {};
  const size_t abase = (size_t)mt * 16 * Kd;
  const size_t nb0 = (size_t)ng * 64 * Kd;
  for (int k0 = 0; k0 < Kd; k0 += 32) {
    v16h a = load_a_frag(A + abase + k0, Kd, lane);
#pragma unroll
    for (int t = 0; t < 4; ++t) {
      v16h b = load_b_frag(Wp + nb0 + (size_t)t * 16 * Kd + k0, Kd, lane);
      acc[t] = wmma_f16(a, b, acc[t]);
    }
  }
#pragma unroll
  for (int t = 0; t < 4; ++t)
    store_tile(acc[t], mt * 16, ng * 64 + t * 16, lane, Ndim, bp, resid,
               rowscale, colvec, colbias, Cf, Ch, act);
}

// ---------------- fused cross-attention: one wave per (b, head, 16 q-rows) --
__global__ __launch_bounds__(64) void k_attn(
    const _Float16* __restrict__ qh, const _Float16* __restrict__ khp,
    const _Float16* __restrict__ vT, const float* __restrict__ srcxy,
    const float* __restrict__ obst, _Float16* __restrict__ outh) {
  __shared__ float s_sc[2][16 * CSP];      // 34,816 B
  __shared__ _Float16 s_pb[2][16 * CSPK];  // 18,432 B  (total 53,248 B)
  const int lane = threadIdx.x, wv = threadIdx.y;
  const int gw = blockIdx.x * 2 + wv;
  const int kt = gw & 255;
  const int h = (gw >> 8) & 3;
  const int b = gw >> 10;
  const int k0 = kt << 4;
  const v16h qa = load_a_frag(qh + ((size_t)(b * CK + k0)) * CD + h * CDH, CD, lane);
  float* sc = s_sc[wv];
  const float scale = 0.17677669529663687f;  // 1/sqrt(32)
  const int n = lane & 15;
  const int mb = (lane >> 4) << 3;
  for (int st = 0; st < 17; ++st) {
    v16h kb = load_b_frag(khp + ((size_t)b * CSP + st * 16) * CD + h * CDH, CD, lane);
    v8f c = {};
    c = wmma_f16(qa, kb, c);
    const int s = st * 16 + n;
    const bool valid = s < CS;
    float sx = 0.f, sy = 0.f;
    if (valid) { sx = srcxy[(b * CS + s) * 2]; sy = srcxy[(b * CS + s) * 2 + 1]; }
#pragma unroll
    for (int r = 0; r < 8; ++r) {
      const int m = mb + r;
      float v = -3.0e38f;
      if (valid) {
        const int kk = k0 + m;
        const float gx = ((kk & 63) + 0.5f) * (1.f / 64.f);
        const float gy = ((kk >> 6) + 0.5f) * (1.f / 64.f);
        const float dx = gx - sx, dy = gy - sy;
        float bias = -sqrtf(dx * dx + dy * dy);
        if (s >= CN) bias -= obst[((size_t)(b * CK + kk)) * CM + (s - CN)];
        v = c[r] * scale + bias;
      }
      sc[m * CSP + s] = v;
    }
  }
  __syncthreads();
  if (lane < 16) {  // one lane per q-row: softmax over S
    float* row = sc + lane * CSP;
    float mx = -3.0e38f;
    for (int s = 0; s < CSP; ++s) mx = fmaxf(mx, row[s]);
    float sum = 0.f;
    for (int s = 0; s < CSP; ++s) { float e = expf(row[s] - mx); row[s] = e; sum += e; }
    const float inv = 1.f / sum;
    _Float16* pr = s_pb[wv] + lane * CSPK;
    for (int s = 0; s < CSP; ++s) pr[s] = (_Float16)(row[s] * inv);
    for (int s = CSP; s < CSPK; ++s) pr[s] = (_Float16)0.f;
  }
  __syncthreads();
  v8f o0 = {}, o1 = {};
  const _Float16* pb = s_pb[wv];
  const _Float16* vb = vT + ((size_t)(b * NH + h)) * CDH * CSPK;
  for (int kc = 0; kc < CSPK; kc += 32) {
    v16h pa = load_a_frag(pb + kc, CSPK, lane);
    v16h b0 = load_b_frag(vb + kc, CSPK, lane);
    v16h b1 = load_b_frag(vb + (size_t)16 * CSPK + kc, CSPK, lane);
    o0 = wmma_f16(pa, b0, o0);
    o1 = wmma_f16(pa, b1, o1);
  }
#pragma unroll
  for (int r = 0; r < 8; ++r) {
    const size_t rowi = (size_t)(b * CK + k0 + mb + r) * CD + h * CDH;
    outh[rowi + n] = (_Float16)o0[r];
    outh[rowi + 16 + n] = (_Float16)o1[r];
  }
}

// ---------------- small VALU kernels ----------------------------------------
__global__ void k_wt(const float* __restrict__ w, _Float16* __restrict__ wt,
                     int din, int dout) {  // Wt[o][i] = w[i][o]
  const int t = blockIdx.x * 256 + threadIdx.x;
  if (t >= din * dout) return;
  const int o = t / din, i = t - o * din;
  wt[t] = (_Float16)w[(size_t)i * dout + o];
}
__global__ void k_cast_h(const float* __restrict__ s, _Float16* __restrict__ d, int n) {
  const int t = blockIdx.x * 256 + threadIdx.x;
  if (t < n) d[t] = (_Float16)s[t];
}
__global__ void k_scalars(const float* a, const float* b, float* s) {
  if (threadIdx.x == 0) { s[0] = log1pf(expf(a[0])); s[1] = log1pf(expf(b[0])); }
}
__global__ void k_cityg(const float* __restrict__ city, float* __restrict__ cg) {
  const int t = blockIdx.x * 256 + threadIdx.x;
  if (t >= CB * CK) return;
  const int k = t & (CK - 1), b = t / CK;
  const int y = k >> 6, x = k & 63;
  cg[t] = city[(size_t)b * 65536 + (y * 4) * 256 + x * 4];
}
__global__ void k_srcxy(const float* __restrict__ mxy, const float* __restrict__ bxy,
                        float* __restrict__ sxy) {
  const int t = blockIdx.x * 256 + threadIdx.x;
  if (t >= CB * CS) return;
  const int s = t % CS, b = t / CS;
  float x, y;
  if (s < CN) { x = mxy[(b * CN + s) * 2]; y = mxy[(b * CN + s) * 2 + 1]; }
  else { const int m = s - CN; x = bxy[(b * CM + m) * 2]; y = bxy[(b * CM + m) * 2 + 1]; }
  sxy[2 * t] = x; sxy[2 * t + 1] = y;
}
__global__ void k_density(const float* __restrict__ mxy, float* __restrict__ dens) {
  const int t = blockIdx.x * 256 + threadIdx.x;
  if (t >= CB * CK) return;
  const int k = t & (CK - 1), b = t / CK;
  const float gx = ((k & 63) + 0.5f) * (1.f / 64.f);
  const float gy = ((k >> 6) + 0.5f) * (1.f / 64.f);
  const float* p = mxy + (size_t)b * CN * 2;
  float acc = 0.f;
  for (int i = 0; i < CN; ++i) {
    const float dx = gx - p[2 * i], dy = gy - p[2 * i + 1];
    acc += expf(-(dx * dx + dy * dy) * 78.125f);  // 1/(2*0.08^2)
  }
  dens[t] = acc * (1.f / CN);
}
__global__ void k_obst(const float* __restrict__ bxy, const float* __restrict__ cg,
                       float* __restrict__ obst) {
  const int t = blockIdx.x * 256 + threadIdx.x;
  if (t >= CB * CK * CM) return;
  const int m = t & (CM - 1);
  const int k = (t / CM) & (CK - 1);
  const int b = t / (CM * CK);
  const float gx = ((k & 63) + 0.5f) * (1.f / 64.f);
  const float gy = ((k >> 6) + 0.5f) * (1.f / 64.f);
  const float bx = bxy[(b * CM + m) * 2], by = bxy[(b * CM + m) * 2 + 1];
  float s = 0.f;
  for (int j = 1; j <= 14; ++j) {
    const float a = j * (1.f / 15.f);
    const float lx = (1.f - a) * gx + a * bx;
    const float ly = (1.f - a) * gy + a * by;
    int xi = (int)rintf(lx * 63.f); xi = xi < 0 ? 0 : (xi > 63 ? 63 : xi);
    int yi = (int)rintf(ly * 63.f); yi = yi < 0 ? 0 : (yi > 63 ? 63 : yi);
    s += cg[b * CK + yi * 64 + xi];
  }
  s *= (1.f / 14.f);
  obst[t] = s < 0.f ? 0.f : (s > 1.f ? 1.f : s);
}
__global__ void k_meas_hidden(const float* __restrict__ mxy, const float* __restrict__ mv,
                              const float* __restrict__ w, const float* __restrict__ bb,
                              _Float16* __restrict__ out) {
  const int r = blockIdx.x * 256 + threadIdx.x;
  if (r >= CB * CN) return;
  const float x0 = mxy[2 * r], x1 = mxy[2 * r + 1], x2 = mv[r];
  _Float16* o = out + (size_t)r * CD;
  for (int j = 0; j < CD; ++j)
    o[j] = (_Float16)gelu_f(x0 * w[j] + x1 * w[CD + j] + x2 * w[2 * CD + j] + bb[j]);
}
__global__ void k_bs_hidden(const float* __restrict__ bxy, const float* __restrict__ w,
                            const float* __restrict__ bb, _Float16* __restrict__ out) {
  const int r = blockIdx.x * 32 + threadIdx.x;
  if (r >= CB * CM) return;
  const float x0 = bxy[2 * r], x1 = bxy[2 * r + 1];
  _Float16* o = out + (size_t)r * CD;
  for (int j = 0; j < CD; ++j)
    o[j] = (_Float16)gelu_f(x0 * w[j] + x1 * w[CD + j] + bb[j]);
}
__global__ void k_build_src(const float* __restrict__ tmp, float* __restrict__ src) {
  const int t = blockIdx.x * 256 + threadIdx.x;
  if (t >= CB * CS * CD) return;
  const int d = t & (CD - 1);
  const int s = (t / CD) % CS;
  const int b = t / (CD * CS);
  src[t] = (s < CN) ? tmp[((size_t)(b * CN + s)) * CD + d]
                    : tmp[(size_t)CB * CN * CD + ((size_t)(b * CM + (s - CN))) * CD + d];
}
__global__ void k_conv1(const float* __restrict__ cg, const float* __restrict__ w,
                        const float* __restrict__ bb, float* __restrict__ h1) {
  const int t = blockIdx.x * 256 + threadIdx.x;
  if (t >= CB * CK * CE) return;
  const int c = t & (CE - 1);
  const int x = (t / CE) & 63;
  const int y = (t / (CE * 64)) & 63;
  const int b = t / (CE * CK);
  float acc = bb[c];
  const float* wc = w + c * 9;
  for (int ky = 0; ky < 3; ++ky) {
    const int yy = y + ky - 1; if (yy < 0 || yy > 63) continue;
    for (int kx = 0; kx < 3; ++kx) {
      const int xx = x + kx - 1; if (xx < 0 || xx > 63) continue;
      acc += cg[b * CK + yy * 64 + xx] * wc[ky * 3 + kx];
    }
  }
  h1[t] = gelu_f(acc);  // channel-last [b][y][x][c]
}
__global__ void k_conv2(const float* __restrict__ h1, const float* __restrict__ w,
                        const float* __restrict__ bb, _Float16* __restrict__ h2) {
  const int t = blockIdx.x * 256 + threadIdx.x;
  if (t >= CB * CK * CE) return;
  const int c = t & (CE - 1);
  const int x = (t / CE) & 63;
  const int y = (t / (CE * 64)) & 63;
  const int b = t / (CE * CK);
  float acc = bb[c];
  for (int ky = 0; ky < 3; ++ky) {
    const int yy = y + ky - 1; if (yy < 0 || yy > 63) continue;
    for (int kx = 0; kx < 3; ++kx) {
      const int xx = x + kx - 1; if (xx < 0 || xx > 63) continue;
      const float* hr = h1 + ((size_t)((b * 64 + yy) * 64 + xx)) * CE;
      const float* wk = w + c * CE * 9 + ky * 3 + kx;
      for (int ci = 0; ci < CE; ++ci) acc += hr[ci] * wk[ci * 9];
    }
  }
  h2[t] = (_Float16)gelu_f(acc);
}
__global__ void k_taskvec(const int* __restrict__ tid, const float* __restrict__ te,
                          const float* __restrict__ tpw, const float* __restrict__ tpb,
                          float* __restrict__ tv) {
  const int t = blockIdx.x * 256 + threadIdx.x;
  if (t >= CB * CD) return;
  const int d = t & (CD - 1), b = t / CD;
  const float* e = te + (size_t)tid[b] * CD;
  float acc = tpb[d];
  for (int i = 0; i < CD; ++i) acc += e[i] * tpw[(size_t)i * CD + d];
  tv[t] = acc;
}
__global__ void k_init_grid(const float* __restrict__ gp, const float* __restrict__ tv,
                            const float* __restrict__ eg, const float* __restrict__ scal,
                            float* __restrict__ grid) {
  const int t = blockIdx.x * 256 + threadIdx.x;
  if (t >= CB * CK * CD) return;
  const int d = t & (CD - 1);
  const int k = (t / CD) & (CK - 1);
  const int b = t / (CD * CK);
  grid[t] = gp[(size_t)k * CD + d] + tv[b * CD + d] + scal[0] * eg[t];
}
__global__ void k_axpy_env(float* __restrict__ grid, const float* __restrict__ eg,
                           const float* __restrict__ scal) {
  const int t = blockIdx.x * 256 + threadIdx.x;
  if (t < CB * CK * CD) grid[t] += scal[1] * eg[t];
}
__global__ void k_layernorm_h(const float* __restrict__ X, const float* __restrict__ g,
                              const float* __restrict__ bb, _Float16* __restrict__ Y,
                              int rows) {
  const int r = blockIdx.x * 256 + threadIdx.x;
  if (r >= rows) return;
  const float* x = X + (size_t)r * CD;
  float m = 0.f;
  for (int i = 0; i < CD; ++i) m += x[i];
  m *= (1.f / CD);
  float v = 0.f;
  for (int i = 0; i < CD; ++i) { const float d = x[i] - m; v += d * d; }
  const float is = rsqrtf(v * (1.f / CD) + 1e-5f);
  _Float16* y = Y + (size_t)r * CD;
  for (int i = 0; i < CD; ++i) y[i] = (_Float16)((x[i] - m) * is * g[i] + bb[i]);
}
__global__ void k_pack_kh(const _Float16* __restrict__ khf, _Float16* __restrict__ khp) {
  const int t = blockIdx.x * 256 + threadIdx.x;
  if (t >= CB * CSP * CD) return;
  const int d = t & (CD - 1);
  const int s = (t / CD) % CSP;
  const int b = t / (CD * CSP);
  khp[t] = (s < CS) ? khf[((size_t)(b * CS + s)) * CD + d] : (_Float16)0.f;
}
__global__ void k_pack_vT(const _Float16* __restrict__ vhf, _Float16* __restrict__ vT) {
  const int t = blockIdx.x * 256 + threadIdx.x;
  if (t >= CB * NH * CDH * CSPK) return;
  const int s = t % CSPK;
  const int d = (t / CSPK) & (CDH - 1);
  const int h = (t / (CSPK * CDH)) & (NH - 1);
  const int b = t / (CSPK * CDH * NH);
  vT[t] = (s < CS) ? vhf[((size_t)(b * CS + s)) * CD + h * CDH + d] : (_Float16)0.f;
}
__global__ void k_chd(const _Float16* __restrict__ f, const float* __restrict__ cw,
                      const float* __restrict__ cb, float* __restrict__ out) {
  const int t = blockIdx.x * 256 + threadIdx.x;
  if (t >= CB * CK) return;
  const _Float16* x = f + (size_t)t * CD;
  float acc = cb[0];
  for (int d = 0; d < CD; ++d) acc += (float)x[d] * cw[d];
  out[t] = acc;
}

// ---------------- host-side helpers -----------------------------------------
static void gemm(hipStream_t st, const _Float16* A, const _Float16* Wt,
                 const float* bias, const float* resid, const float* rowscale,
                 const float* colvec, const float* colbias, float* Cf,
                 _Float16* Ch, int Mrows, int Ndim, int Kd, int act,
                 const int* task = nullptr, int rpt = 1) {
  const int tiles = (Mrows / 16) * (Ndim / 64);
  k_gemm<<<(tiles + 7) / 8, dim3(32, 8), 0, st>>>(A, Wt, bias, resid, rowscale,
                                                  colvec, colbias, Cf, Ch, Mrows,
                                                  Ndim, Kd, act, task, rpt);
}
static void conv_wt(hipStream_t st, const float* w, _Float16* wt, int din, int dout) {
  const int n = din * dout;
  k_wt<<<(n + 255) / 256, 256, 0, st>>>(w, wt, din, dout);
}

extern "C" void kernel_launch(void* const* d_in, const int* in_sizes, int n_in,
                              void* d_out, int out_size, void* d_ws, size_t ws_size,
                              hipStream_t stream) {
  (void)in_sizes; (void)n_in; (void)out_size; (void)ws_size;
  const float* meas_xy = (const float*)d_in[0];
  const float* meas_v  = (const float*)d_in[1];
  const float* bs_xy   = (const float*)d_in[2];
  const float* city    = (const float*)d_in[3];
  const int*   task_id = (const int*)d_in[4];
  // params flattened as a jax pytree (dict keys sorted alphabetically)
  int p = 5;
  struct BW {
    const float *dens_b, *dens_w, *ff1_b, *ff1_w, *ff2_b, *ff2_w,
                *lnf_b, *lnf_g, *lnkv_b, *lnkv_g, *lnq_b, *lnq_g,
                *wk_b, *wk_w, *wo_b, *wo_w, *wq_b, *wq_w, *wv_b, *wv_w;
  } bw[2];
  for (int li = 0; li < 2; ++li) {
    BW& w = bw[li];
    w.dens_b = (const float*)d_in[p++]; w.dens_w = (const float*)d_in[p++];
    w.ff1_b  = (const float*)d_in[p++]; w.ff1_w  = (const float*)d_in[p++];
    w.ff2_b  = (const float*)d_in[p++]; w.ff2_w  = (const float*)d_in[p++];
    w.lnf_b  = (const float*)d_in[p++]; w.lnf_g  = (const float*)d_in[p++];
    w.lnkv_b = (const float*)d_in[p++]; w.lnkv_g = (const float*)d_in[p++];
    w.lnq_b  = (const float*)d_in[p++]; w.lnq_g  = (const float*)d_in[p++];
    w.wk_b   = (const float*)d_in[p++]; w.wk_w   = (const float*)d_in[p++];
    w.wo_b   = (const float*)d_in[p++]; w.wo_w   = (const float*)d_in[p++];
    w.wq_b   = (const float*)d_in[p++]; w.wq_w   = (const float*)d_in[p++];
    w.wv_b   = (const float*)d_in[p++]; w.wv_w   = (const float*)d_in[p++];
  }
  const float* bs1_b = (const float*)d_in[p++]; const float* bs1_w = (const float*)d_in[p++];
  const float* bs2_b = (const float*)d_in[p++]; const float* bs2_w = (const float*)d_in[p++];
  const float* chd_b = (const float*)d_in[p++]; const float* chd_w = (const float*)d_in[p++];
  const float* s1_raw = (const float*)d_in[p++];                      // env_block0_scale_raw
  const float* c1_b = (const float*)d_in[p++]; const float* c1_w = (const float*)d_in[p++];
  const float* c2_b = (const float*)d_in[p++]; const float* c2_w = (const float*)d_in[p++];
  const float* c3_b = (const float*)d_in[p++]; const float* c3_w = (const float*)d_in[p++];
  const float* g1_b = (const float*)d_in[p++]; const float* g1_w = (const float*)d_in[p++];
  const float* g2_b = (const float*)d_in[p++]; const float* g2_w = (const float*)d_in[p++];
  const float* s0_raw = (const float*)d_in[p++];                      // env_init_scale_raw
  const float* grid_pos = (const float*)d_in[p++];
  const float* m1_b = (const float*)d_in[p++]; const float* m1_w = (const float*)d_in[p++];
  const float* m2_b = (const float*)d_in[p++]; const float* m2_w = (const float*)d_in[p++];
  const float* temb = (const float*)d_in[p++];
  const float* tp_b = (const float*)d_in[p++]; const float* tp_w = (const float*)d_in[p++];
  const float* tfp_b = (const float*)d_in[p++]; const float* tfp_w = (const float*)d_in[p++];

  // bump allocator over workspace
  char* base = (char*)d_ws;
  size_t off = 0;
  auto alloc = [&](size_t bytes) -> void* {
    void* r = base + off;
    off = (off + bytes + 255) & ~(size_t)255;
    return r;
  };
  float* grid = (float*)alloc((size_t)CB * CK * CD * 4);
  float* envg = (float*)alloc((size_t)CB * CK * CD * 4);
  float* tmp  = (float*)alloc((size_t)CB * CK * CD * 4);
  float* src  = (float*)alloc((size_t)CB * CS * CD * 4);
  _Float16* acth = (_Float16*)alloc((size_t)CB * CK * CD * 2);
  _Float16* act2 = (_Float16*)alloc((size_t)CB * CK * CD * 2);
  _Float16* ffh  = (_Float16*)alloc((size_t)CB * CK * CD4 * 2);
  _Float16* kvnh = (_Float16*)alloc((size_t)CB * CS * CD * 2);
  _Float16* khf  = (_Float16*)alloc((size_t)CB * CS * CD * 2);
  _Float16* vhf  = (_Float16*)alloc((size_t)CB * CS * CD * 2);
  _Float16* khp  = (_Float16*)alloc((size_t)CB * CSP * CD * 2);
  _Float16* vT   = (_Float16*)alloc((size_t)CB * NH * CDH * CSPK * 2);
  float* cityg = (float*)alloc((size_t)CB * CK * 4);
  float* h1    = (float*)alloc((size_t)CB * CK * CE * 4);
  _Float16* h2h = (_Float16*)alloc((size_t)CB * CK * CE * 2);
  float* obst = (float*)alloc((size_t)CB * CK * CM * 4);
  float* dens = (float*)alloc((size_t)CB * CK * 4);
  float* sxy  = (float*)alloc((size_t)CB * CS * 2 * 4);
  float* tvec = (float*)alloc((size_t)CB * CD * 4);
  float* scal = (float*)alloc(2 * 4);
  _Float16* m2t = (_Float16*)alloc(16384 * 2);
  _Float16* b2t = (_Float16*)alloc(16384 * 2);
  _Float16* c3h = (_Float16*)alloc(4096 * 2);
  _Float16* g1t = (_Float16*)alloc(16384 * 2);
  _Float16* g2t = (_Float16*)alloc(16384 * 2);
  _Float16* tfpt = (_Float16*)alloc(4 * 16384 * 2);
  _Float16 *wqt[2], *wkt[2], *wvt[2], *wot[2], *f1t[2], *f2t[2];
  for (int li = 0; li < 2; ++li) {
    wqt[li] = (_Float16*)alloc(16384 * 2);
    wkt[li] = (_Float16*)alloc(16384 * 2);
    wvt[li] = (_Float16*)alloc(16384 * 2);
    wot[li] = (_Float16*)alloc(16384 * 2);
    f1t[li] = (_Float16*)alloc((size_t)CD * CD4 * 2);
    f2t[li] = (_Float16*)alloc((size_t)CD * CD4 * 2);
  }

  // ---- weight conversion (f32 -> f16, [dout][din]) ----
  conv_wt(stream, m2_w, m2t, CD, CD);
  conv_wt(stream, bs2_w, b2t, CD, CD);
  k_cast_h<<<(4096 + 255) / 256, 256, 0, stream>>>(c3_w, c3h, 4096);  // (128,32,1,1) already [N][K]
  conv_wt(stream, g1_w, g1t, CD, CD);
  conv_wt(stream, g2_w, g2t, CD, CD);
  for (int li = 0; li < 2; ++li) {
    conv_wt(stream, bw[li].wq_w, wqt[li], CD, CD);
    conv_wt(stream, bw[li].wk_w, wkt[li], CD, CD);
    conv_wt(stream, bw[li].wv_w, wvt[li], CD, CD);
    conv_wt(stream, bw[li].wo_w, wot[li], CD, CD);
    conv_wt(stream, bw[li].ff1_w, f1t[li], CD, CD4);
    conv_wt(stream, bw[li].ff2_w, f2t[li], CD4, CD);
  }
  for (int t = 0; t < 4; ++t)
    conv_wt(stream, tfp_w + (size_t)t * CD * CD, tfpt + (size_t)t * CD * CD, CD, CD);

  // ---- precompute ----
  k_scalars<<<1, 1, 0, stream>>>(s0_raw, s1_raw, scal);
  k_cityg<<<(CB * CK + 255) / 256, 256, 0, stream>>>(city, cityg);
  k_srcxy<<<(CB * CS + 255) / 256, 256, 0, stream>>>(meas_xy, bs_xy, sxy);
  k_density<<<(CB * CK + 255) / 256, 256, 0, stream>>>(meas_xy, dens);
  k_obst<<<(CB * CK * CM + 255) / 256, 256, 0, stream>>>(bs_xy, cityg, obst);

  // ---- source tokens ----
  k_meas_hidden<<<(CB * CN + 255) / 256, 256, 0, stream>>>(meas_xy, meas_v, m1_w, m1_b, acth);
  k_bs_hidden<<<1, 32, 0, stream>>>(bs_xy, bs1_w, bs1_b, acth + (size_t)CB * CN * CD);
  gemm(stream, acth, m2t, m2_b, nullptr, nullptr, nullptr, nullptr, tmp, nullptr,
       CB * CN, CD, CD, 0);
  gemm(stream, acth + (size_t)CB * CN * CD, b2t, bs2_b, nullptr, nullptr, nullptr,
       nullptr, tmp + (size_t)CB * CN * CD, nullptr, CB * CM, CD, CD, 0);
  k_build_src<<<(CB * CS * CD + 255) / 256, 256, 0, stream>>>(tmp, src);

  // ---- env CNN + env MLP ----
  k_conv1<<<(CB * CK * CE + 255) / 256, 256, 0, stream>>>(cityg, c1_w, c1_b, h1);
  k_conv2<<<(CB * CK * CE + 255) / 256, 256, 0, stream>>>(h1, c2_w, c2_b, h2h);
  gemm(stream, h2h, c3h, c3_b, nullptr, nullptr, nullptr, nullptr, nullptr, acth,
       CB * CK, CD, CE, 0);                                      // env feature map (f16)
  gemm(stream, acth, g1t, g1_b, nullptr, nullptr, nullptr, nullptr, nullptr, act2,
       CB * CK, CD, CD, 1);                                      // gelu
  gemm(stream, act2, g2t, g2_b, nullptr, nullptr, nullptr, nullptr, envg, nullptr,
       CB * CK, CD, CD, 0);

  // ---- init grid ----
  k_taskvec<<<(CB * CD + 255) / 256, 256, 0, stream>>>(task_id, temb, tp_w, tp_b, tvec);
  k_init_grid<<<(CB * CK * CD + 255) / 256, 256, 0, stream>>>(grid_pos, tvec, envg, scal, grid);

  // ---- transformer blocks ----
  for (int li = 0; li < 2; ++li) {
    const BW& w = bw[li];
    k_layernorm_h<<<(CB * CK + 255) / 256, 256, 0, stream>>>(grid, w.lnq_g, w.lnq_b, acth, CB * CK);
    k_layernorm_h<<<(CB * CS + 255) / 256, 256, 0, stream>>>(src, w.lnkv_g, w.lnkv_b, kvnh, CB * CS);
    gemm(stream, acth, wqt[li], w.wq_b, nullptr, nullptr, nullptr, nullptr, nullptr,
         act2, CB * CK, CD, CD, 0);                               // qh (f16)
    gemm(stream, kvnh, wkt[li], w.wk_b, nullptr, nullptr, nullptr, nullptr, nullptr,
         khf, CB * CS, CD, CD, 0);
    gemm(stream, kvnh, wvt[li], w.wv_b, nullptr, nullptr, nullptr, nullptr, nullptr,
         vhf, CB * CS, CD, CD, 0);
    k_pack_kh<<<(CB * CSP * CD + 255) / 256, 256, 0, stream>>>(khf, khp);
    k_pack_vT<<<(CB * NH * CDH * CSPK + 255) / 256, 256, 0, stream>>>(vhf, vT);
    k_attn<<<CB * NH * (CK / 16) / 2, dim3(32, 2), 0, stream>>>(act2, khp, vT, sxy, obst, acth);
    gemm(stream, acth, wot[li], w.wo_b, grid, dens, w.dens_w, w.dens_b, grid, nullptr,
         CB * CK, CD, CD, 0);                                     // grid += wo(out)+dens
    k_layernorm_h<<<(CB * CK + 255) / 256, 256, 0, stream>>>(grid, w.lnf_g, w.lnf_b, acth, CB * CK);
    gemm(stream, acth, f1t[li], w.ff1_b, nullptr, nullptr, nullptr, nullptr, nullptr,
         ffh, CB * CK, CD4, CD, 1);                               // gelu, f16 hidden
    gemm(stream, ffh, f2t[li], w.ff2_b, grid, nullptr, nullptr, nullptr, grid, nullptr,
         CB * CK, CD, CD4, 0);                                    // grid += ff2
    if (li == 0)
      k_axpy_env<<<(CB * CK * CD + 255) / 256, 256, 0, stream>>>(grid, envg, scal);
  }

  // ---- task-filtered projection + channel head ----
  k_cast_h<<<(CB * CK * CD + 255) / 256, 256, 0, stream>>>(grid, acth, CB * CK * CD);
  gemm(stream, acth, tfpt, tfp_b, nullptr, nullptr, nullptr, nullptr, nullptr, act2,
       CB * CK, CD, CD, 1, task_id, CK);                          // gelu(grid @ w[task])
  k_chd<<<(CB * CK + 255) / 256, 256, 0, stream>>>(act2, chd_w, chd_b, (float*)d_out);
}
